// PredNet_hGRU_67370857005221
// MI455X (gfx1250) — matrix operand order repaired and checked by reading, not compiled, not run on hardware
//
#include <hip/hip_runtime.h>
#include <hip/hip_bf16.h>
#include <math.h>

// ---------------------------------------------------------------- types ----
typedef __attribute__((ext_vector_type(16))) __bf16 bf16x16;
typedef __attribute__((ext_vector_type(8)))  float  f32x8;

union Frag16 {
  bf16x16 v;
  __bf16 bf[16];
  unsigned short h[16];
  uint4 q[2];
};

static constexpr int P16c(int c) { return (c + 15) & ~15; }
static constexpr int P32c(int c) { return (c + 31) & ~31; }

__device__ __forceinline__ float sigm_f(float x) { return 1.f / (1.f + __expf(-x)); }
__device__ __forceinline__ float sp_f(float x)   { return (x > 20.f) ? x : log1pf(__expf(x)); }

enum { EP_LIN = 0, EP_SIG = 1, EP_SIGMUL = 2, EP_RELU = 3, EP_RELUCLIP = 4, EP_SP = 5 };

#define GS_LOOP(i, total) \
  for (int i = blockIdx.x * blockDim.x + threadIdx.x; i < (total); i += gridDim.x * blockDim.x)

// ------------------------------------------------------- WMMA conv (core) --
// in : [B][H][W][Cis] f32 (NHWC, channel-padded)  out: [B][H][W][Cos] f32
// Wb : bf16 [Coutp][TAPS][CINP], zero outside real cout/cin
// Each wave computes a 16(cout) x 32(positions) tile: one A fragment shared
// by two B fragments -> two v_wmma per K-chunk; K loop fully unrolled.
template <int TAPS, int CINP, bool GUARD>
__global__ __launch_bounds__(128) void k_conv_wmma(
    const float* __restrict__ in, float* __restrict__ out,
    const unsigned short* __restrict__ Wb, const float* __restrict__ bias,
    const float* __restrict__ aux,
    int H, int W, int Cis, int Cos, int coutReal, int ep, int totalTiles)
{
  const int lane = threadIdx.x & 31;
  const int half = lane >> 4;
  const int l15  = lane & 15;
  const int Mtiles = Cos >> 4;
  const int wave0  = blockIdx.x * (blockDim.x >> 5) + (threadIdx.x >> 5);
  const int nWaves = gridDim.x * (blockDim.x >> 5);

  for (int tile = wave0; tile < totalTiles; tile += nWaves) {
    const int mt = tile % Mtiles;
    const int sp = tile / Mtiles;
    const int x0 = (sp << 5) % W;
    const int y  = ((sp << 5) / W) % H;
    const int b  = (sp << 5) / (W * H);
    const int m0 = mt << 4;

    f32x8 acc0, acc1;
#pragma unroll
    for (int j = 0; j < 8; ++j) {
      int m = m0 + j + half * 8;
      float bv = (bias != nullptr && m < coutReal) ? bias[m] : 0.f;
      acc0[j] = bv;
      acc1[j] = bv;
    }

    const int mRow = m0 + l15;            // A row (output channel)
    const int xn0  = x0 + l15;            // B columns (two spatial sub-tiles)
    const int xn1  = x0 + 16 + l15;
    const unsigned short* apR = Wb + (size_t)(mRow * TAPS) * CINP;

    for (int tap = 0; tap < TAPS; ++tap) {
      const int dy = (TAPS == 9) ? (tap / 3 - 1) : 0;
      const int dx = (TAPS == 9) ? (tap % 3 - 1) : 0;
      const int yy = y + dy;
      const bool rowok = (yy >= 0) & (yy < H);
      const int xx0 = xn0 + dx;
      const int xx1 = xn1 + dx;
      const bool ok0 = rowok & (xx0 >= 0) & (xx0 < W);
      const bool ok1 = rowok & (xx1 >= 0) & (xx1 < W);
      const float* bp0 = in + (size_t)((b * H + yy) * W + xx0) * Cis;
      const float* bp1 = in + (size_t)((b * H + yy) * W + xx1) * Cis;
      const unsigned short* ap = apR + (size_t)tap * CINP;
      if (tap + 1 < TAPS) __builtin_prefetch(ap + CINP, 0, 2);

#pragma unroll
      for (int ci0 = 0; ci0 < CINP; ci0 += 32) {
        Frag16 a, b0, b1;
        // A: lane half h holds 16 consecutive K (ci0 + 16h .. +15), matching B
        a.q[0] = *(const uint4*)(ap + ci0 + half * 16);
        a.q[1] = *(const uint4*)(ap + ci0 + half * 16 + 8);
        // B: lane = column, elements = 16 consecutive K (channels).
        // cb and Cis are multiples of 16 -> whole-chunk bounds test.
        const int cb = ci0 + half * 16;
        const bool cok = !GUARD || (cb < Cis);
        const float4 z4 = make_float4(0.f, 0.f, 0.f, 0.f);
        const float4* q0 = (const float4*)(bp0 + cb);
        const float4* q1 = (const float4*)(bp1 + cb);
        float4 f0[4], f1[4];
#pragma unroll
        for (int k = 0; k < 4; ++k) {
          f0[k] = (ok0 && cok) ? q0[k] : z4;
          f1[k] = (ok1 && cok) ? q1[k] : z4;
        }
#pragma unroll
        for (int k = 0; k < 4; ++k) {
          b0.bf[4 * k + 0] = (__bf16)f0[k].x; b0.bf[4 * k + 1] = (__bf16)f0[k].y;
          b0.bf[4 * k + 2] = (__bf16)f0[k].z; b0.bf[4 * k + 3] = (__bf16)f0[k].w;
          b1.bf[4 * k + 0] = (__bf16)f1[k].x; b1.bf[4 * k + 1] = (__bf16)f1[k].y;
          b1.bf[4 * k + 2] = (__bf16)f1[k].z; b1.bf[4 * k + 3] = (__bf16)f1[k].w;
        }
        acc0 = __builtin_amdgcn_wmma_f32_16x16x32_bf16(false, a.v, false, b0.v,
                                                       (short)0, acc0, false, false);
        acc1 = __builtin_amdgcn_wmma_f32_16x16x32_bf16(false, a.v, false, b1.v,
                                                       (short)0, acc1, false, false);
      }
    }

    float* op0 = out + (size_t)((b * H + y) * W + xn0) * Cos;
    float* op1 = out + (size_t)((b * H + y) * W + xn1) * Cos;
    const float* xq0 = (aux != nullptr) ? (aux + (size_t)((b * H + y) * W + xn0) * Cos) : nullptr;
    const float* xq1 = (aux != nullptr) ? (aux + (size_t)((b * H + y) * W + xn1) * Cos) : nullptr;
#pragma unroll
    for (int j = 0; j < 8; ++j) {
      int m = m0 + j + half * 8;
      float v0 = acc0[j], v1 = acc1[j];
      if (ep == EP_SIG)           { v0 = sigm_f(v0); v1 = sigm_f(v1); }
      else if (ep == EP_SIGMUL)   { v0 = sigm_f(v0) * xq0[m]; v1 = sigm_f(v1) * xq1[m]; }
      else if (ep == EP_RELU)     { v0 = v0 > 0.f ? v0 : 0.f; v1 = v1 > 0.f ? v1 : 0.f; }
      else if (ep == EP_RELUCLIP) { v0 = v0 > 0.f ? v0 : 0.f; v0 = v0 < 255.f ? v0 : 255.f;
                                    v1 = v1 > 0.f ? v1 : 0.f; v1 = v1 < 255.f ? v1 : 255.f; }
      else if (ep == EP_SP)       { v0 = sp_f(v0); v1 = sp_f(v1); }
      op0[m] = v0;
      op1[m] = v1;
    }
  }
}

// ------------------------------------------------------- weight packing ----
__global__ void k_packw(const float* __restrict__ w, unsigned short* __restrict__ wb,
                        int Cout, int Cin, int taps, int Cinp, int total)
{
  GS_LOOP(i, total) {
    int ci = i % Cinp;
    int r  = i / Cinp;
    int tap = r % taps;
    int co  = r / taps;
    float v = 0.f;
    if (co < Cout && ci < Cin) v = w[(size_t)(co * Cin + ci) * taps + tap];
    __bf16 bv = (__bf16)v;
    wb[i] = __builtin_bit_cast(unsigned short, bv);
  }
}

// ----------------------------------------------------------- pointwise -----
__global__ void k_fill(float* __restrict__ p, int total, int Cs, int Creal, float v)
{
  GS_LOOP(i, total) p[i] = ((i % Cs) < Creal) ? v : 0.f;
}

__global__ void k_supp(const float* __restrict__ inp, const float* __restrict__ c0,
                       const float* __restrict__ h, const float* __restrict__ alpha,
                       const float* __restrict__ mu, float* __restrict__ out,
                       int total, int Cs, int Creal)
{
  GS_LOOP(i, total) {
    int c = i % Cs;
    float al = (c < Creal) ? alpha[c] : 0.f;
    float m  = (c < Creal) ? mu[c]    : 0.f;
    out[i] = sp_f(sp_f(inp[i]) - sp_f(c0[i] * (al * h[i] + m)));
  }
}

__global__ void k_combine(const float* __restrict__ supp, const float* __restrict__ exc,
                          const float* __restrict__ g2, const float* __restrict__ h,
                          const float* __restrict__ kap, const float* __restrict__ wwv,
                          float* __restrict__ out, int total, int Cs, int Creal)
{
  GS_LOOP(i, total) {
    int c = i % Cs;
    if (c >= Creal) { out[i] = 0.f; }
    else {
      float s = supp[i], e = exc[i], g = g2[i], hh = h[i];
      float ht = sp_f(kap[c] * (s + e) + wwv[c] * s * e);
      out[i] = (1.f - g) * hh + g * ht;
    }
  }
}

__global__ void k_concat(const float* __restrict__ e, const float* __restrict__ r2,
                         float* __restrict__ out, int H, int W, int CsE, int CsR2,
                         int C1, int C2, int CsC, int total)
{
  GS_LOOP(i, total) {
    int c = i % CsC;
    int p = i / CsC;
    int x = p % W;
    int y = (p / W) % H;
    int b = p / (W * H);
    float v = 0.f;
    if (c < C1)           v = e[(size_t)((b * H + y) * W + x) * CsE + c];
    else if (c < C1 + C2) v = r2[(size_t)((b * (H >> 1) + (y >> 1)) * (W >> 1) + (x >> 1)) * CsR2 + (c - C1)];
    out[i] = v;
  }
}

__global__ void k_pool(const float* __restrict__ in, float* __restrict__ out,
                       int Ho, int Wo, int Cs, int total)
{
  GS_LOOP(i, total) {
    int c = i % Cs;
    int p = i / Cs;
    int x = p % Wo;
    int y = (p / Wo) % Ho;
    int b = p / (Wo * Ho);
    int Wi = Wo * 2;
    const float* bp = in + (size_t)((b * (Ho * 2) + 2 * y) * Wi + 2 * x) * Cs + c;
    float v00 = bp[0], v01 = bp[Cs];
    float v10 = bp[(size_t)Wi * Cs], v11 = bp[(size_t)Wi * Cs + Cs];
    out[i] = fmaxf(fmaxf(v00, v01), fmaxf(v10, v11));
  }
}

__global__ void k_xcvt(const float* __restrict__ x, float* __restrict__ out, int t, int total)
{
  GS_LOOP(i, total) {
    int c  = i & 15;
    int p  = i >> 4;
    int xx = p & 255;
    int y  = (p >> 8) & 255;
    int b  = p >> 16;
    float v = 0.f;
    if (c < 3) v = x[(size_t)((b * 3 + c) * 8 + t) * 65536 + (y << 8) + xx];
    out[i] = v;
  }
}

__global__ __launch_bounds__(256) void k_err(
    const float* __restrict__ ahat, const float* __restrict__ a, float* __restrict__ eout,
    float* __restrict__ partial, int HW, int CsE, int CsR, int C, int doFerr)
{
  const int b = blockIdx.y;
  const int total = HW * CsE;
  float se = 0.f, sf = 0.f;
  for (int i = blockIdx.x * blockDim.x + threadIdx.x; i < total; i += gridDim.x * blockDim.x) {
    int c = i % CsE;
    int p = i / CsE;
    size_t posR = ((size_t)b * HW + p) * CsR;
    size_t posE = ((size_t)b * HW + p) * CsE;
    float v = 0.f;
    if (c < C) {
      float d = ahat[posR + c] - a[posR + c];
      v = d > 0.f ? d : 0.f;
      if (doFerr) sf += d * d;
    } else if (c < 2 * C) {
      float d = a[posR + c - C] - ahat[posR + c - C];
      v = d > 0.f ? d : 0.f;
    }
    eout[posE + c] = v;
    se += v;
  }
  __shared__ float sh[256], sh2[256];
  sh[threadIdx.x] = se; sh2[threadIdx.x] = sf;
  __syncthreads();
  for (int s = 128; s > 0; s >>= 1) {
    if ((int)threadIdx.x < s) { sh[threadIdx.x] += sh[threadIdx.x + s]; sh2[threadIdx.x] += sh2[threadIdx.x + s]; }
    __syncthreads();
  }
  if (threadIdx.x == 0) {
    partial[(size_t)(b * gridDim.x + blockIdx.x) * 2 + 0] = sh[0];
    partial[(size_t)(b * gridDim.x + blockIdx.x) * 2 + 1] = sh2[0];
  }
}

__global__ void k_reduce_err(const float* __restrict__ partial, int nblocks,
                             float* __restrict__ meSlot, float* __restrict__ feSlot, int doFerr)
{
  int b = threadIdx.x;
  if (b >= 4) return;
  float se = 0.f, sf = 0.f;
  for (int i = 0; i < nblocks; ++i) {
    se += partial[(size_t)(b * nblocks + i) * 2 + 0];
    sf += partial[(size_t)(b * nblocks + i) * 2 + 1];
  }
  meSlot[b * 4] = se;
  if (doFerr) feSlot[b] = sf;
}

__global__ void k_finalize(const float* __restrict__ me, const float* __restrict__ fe,
                           float* __restrict__ out)
{
  const float wgt[4] = {1.f, 0.1f, 0.1f, 0.1f};
  const float cnt[4] = {6.f * 65536.f, 96.f * 16384.f, 192.f * 4096.f, 384.f * 1024.f};
  int i = threadIdx.x;
  if (i < 28) {
    int b = i / 7, tt = i % 7 + 1;
    float s = 0.f;
    for (int l = 0; l < 4; ++l) s += wgt[l] * me[tt * 16 + b * 4 + l] / cnt[l];
    out[b * 7 + (i % 7)] = s;
  }
  if (i == 31) {
    float s = 0.f;
    for (int t = 1; t < 8; ++t)
      for (int b = 0; b < 4; ++b) s += fe[t * 4 + b];
    out[28] = s / (3.f * 65536.f) / 7.f / 4.f;
  }
}

// ---------------------------------------------------- conv dispatcher ------
static void conv_dispatch(hipStream_t s, const float* in, float* out,
                          const unsigned short* wb, const float* bias, const float* aux,
                          int H, int W, int Cis, int cin, int cout, int taps, int ep)
{
  const int Cinp = P32c(cin), Cos = P16c(cout);
  const bool guard = Cis < Cinp;
  long total = (long)4 * H * W / 32 * (Cos / 16);
  int blocks = (int)((total + 3) / 4);
  if (blocks > 6144) blocks = 6144;
  if (blocks < 1) blocks = 1;
#define LC(T, C, G) k_conv_wmma<T, C, G><<<blocks, 128, 0, s>>>( \
    in, out, wb, bias, aux, H, W, Cis, Cos, cout, ep, (int)total)
  if (taps == 1) {
    if (Cinp == 32)       LC(1, 32, true);
    else if (Cinp == 64)  { if (guard) LC(1, 64, true); else LC(1, 64, false); }
    else if (Cinp == 96)  LC(1, 96, false);
    else if (Cinp == 192) LC(1, 192, false);
    else                  LC(1, 384, false);
  } else {
    if (Cinp == 32)       LC(9, 32, true);
    else if (Cinp == 64)  LC(9, 64, true);
    else if (Cinp == 96)  LC(9, 96, false);
    else                  LC(9, 192, false);
  }
#undef LC
}

// ---------------------------------------------------------------- host -----
extern "C" void kernel_launch(void* const* d_in, const int* in_sizes, int n_in,
                              void* d_out, int out_size, void* d_ws, size_t ws_size,
                              hipStream_t stream)
{
  (void)in_sizes; (void)n_in; (void)out_size; (void)ws_size;
  const int RCc[4]   = {3, 48, 96, 192};
  const int Hl[4]    = {256, 128, 64, 32};
  const int CsR[4]   = {16, 48, 96, 192};
  const int CsE[4]   = {16, 96, 192, 384};
  const int catCin[3] = {54, 192, 384};
  const int CsCat[3]  = {64, 192, 384};
  const int Bn = 4;

  const float* x = (const float*)d_in[0];
  const float *ci_w[4], *ci_b[4], *u0_w[4], *u0_b[4], *u1_w[4], *u1_b[4];
  const float *winh[4], *wexc[4], *alpha[4], *mu[4], *kappa[4], *wwp[4];
  const float *conv_w[4], *conv_b[4], *upd_w[3], *upd_b[3];
  for (int l = 0; l < 4; ++l) {
    ci_w[l]  = (const float*)d_in[1 + l];   ci_b[l]  = (const float*)d_in[5 + l];
    u0_w[l]  = (const float*)d_in[9 + l];   u0_b[l]  = (const float*)d_in[13 + l];
    u1_w[l]  = (const float*)d_in[17 + l];  u1_b[l]  = (const float*)d_in[21 + l];
    winh[l]  = (const float*)d_in[25 + l];  wexc[l]  = (const float*)d_in[29 + l];
    alpha[l] = (const float*)d_in[33 + l];  mu[l]    = (const float*)d_in[37 + l];
    kappa[l] = (const float*)d_in[41 + l];  wwp[l]   = (const float*)d_in[45 + l];
    conv_w[l] = (const float*)d_in[49 + l]; conv_b[l] = (const float*)d_in[53 + l];
  }
  for (int l = 0; l < 3; ++l) { upd_w[l] = (const float*)d_in[57 + l]; upd_b[l] = (const float*)d_in[60 + l]; }

  char* base = (char*)d_ws;
  size_t cur = 0;
  auto alloc = [&](size_t bytes) -> void* {
    void* p = base + cur;
    cur += (bytes + 255) & ~(size_t)255;
    return p;
  };

  long S[4]; for (int l = 0; l < 4; ++l) S[l] = (long)Bn * Hl[l] * Hl[l];
  float *R[4], *Rn[4], *E[4];
  for (int l = 0; l < 4; ++l) R[l]  = (float*)alloc((size_t)S[l] * CsR[l] * 4);
  for (int l = 0; l < 4; ++l) Rn[l] = (float*)alloc((size_t)S[l] * CsR[l] * 4);
  for (int l = 0; l < 4; ++l) E[l]  = (float*)alloc((size_t)S[l] * CsE[l] * 4);
  const size_t tmax = (size_t)S[0] * CsR[0];
  float* T1 = (float*)alloc(tmax * 4);
  float* T2 = (float*)alloc(tmax * 4);
  float* T3 = (float*)alloc(tmax * 4);
  float* T4 = (float*)alloc(tmax * 4);
  float* Tcat = (float*)alloc((size_t)S[0] * 64 * 4);
  float* Tupd = (float*)alloc((size_t)S[0] * 48 * 4);
  float* TA   = (float*)alloc(tmax * 4);
  float* TH   = (float*)alloc(tmax * 4);
  float* me = (float*)alloc(8 * 4 * 4 * sizeof(float));
  float* fe = (float*)alloc(8 * 4 * sizeof(float));
  float* partial = (float*)alloc((size_t)4 * 1024 * 2 * sizeof(float));

  auto wballoc = [&](int cout, int cin, int taps) -> unsigned short* {
    return (unsigned short*)alloc((size_t)P16c(cout) * taps * P32c(cin) * 2);
  };
  auto packw = [&](const float* w, unsigned short* wb, int cout, int cin, int taps) {
    int total = P16c(cout) * taps * P32c(cin);
    int blocks = (total + 255) / 256; if (blocks > 4096) blocks = 4096;
    k_packw<<<blocks, 256, 0, stream>>>(w, wb, cout, cin, taps, P32c(cin), total);
  };

  unsigned short *wb_ci[4], *wb_u0[4], *wb_u1[4], *wb_winh[4], *wb_wexc[4], *wb_conv[4], *wb_upd[3];
  for (int l = 0; l < 4; ++l) {
    int cinci = (l < 3) ? catCin[l] : 384;
    wb_ci[l]   = wballoc(RCc[l], cinci, 1);   packw(ci_w[l],  wb_ci[l],   RCc[l], cinci, 1);
    wb_u0[l]   = wballoc(RCc[l], RCc[l], 1);  packw(u0_w[l],  wb_u0[l],   RCc[l], RCc[l], 1);
    wb_u1[l]   = wballoc(RCc[l], RCc[l], 1);  packw(u1_w[l],  wb_u1[l],   RCc[l], RCc[l], 1);
    wb_winh[l] = wballoc(RCc[l], RCc[l], 9);  packw(winh[l],  wb_winh[l], RCc[l], RCc[l], 9);
    wb_wexc[l] = wballoc(RCc[l], RCc[l], 9);  packw(wexc[l],  wb_wexc[l], RCc[l], RCc[l], 9);
    wb_conv[l] = wballoc(RCc[l], RCc[l], 9);  packw(conv_w[l], wb_conv[l], RCc[l], RCc[l], 9);
  }
  for (int l = 0; l < 3; ++l) {
    wb_upd[l] = wballoc(RCc[l + 1], 2 * RCc[l], 9);
    packw(upd_w[l], wb_upd[l], RCc[l + 1], 2 * RCc[l], 9);
  }

  auto pwb = [](long total) -> int {
    long b = (total + 255) / 256;
    return (int)(b > 16384 ? 16384 : (b < 1 ? 1 : b));
  };
  auto conv = [&](const float* in, float* out, const unsigned short* wb, const float* bias,
                  const float* aux, int Hc, int Wc, int Cis, int cin, int cout, int taps, int ep) {
    conv_dispatch(stream, in, out, wb, bias, aux, Hc, Wc, Cis, cin, cout, taps, ep);
  };

  // ---- initial state: R = ln(2) (real channels), E = 0 ----
  for (int l = 0; l < 4; ++l) {
    long tot = S[l] * (long)CsR[l];
    k_fill<<<pwb(tot), 256, 0, stream>>>(R[l], (int)tot, CsR[l], RCc[l], 0.6931471805599453f);
  }
  for (int l = 0; l < 4; ++l) {
    long tot = S[l] * (long)CsE[l];
    k_fill<<<pwb(tot), 256, 0, stream>>>(E[l], (int)tot, CsE[l], CsE[l], 0.f);
  }

  // ---- recurrence over T = 8 timesteps ----
  for (int t = 0; t < 8; ++t) {
    // top-down
    for (int l = 3; l >= 0; --l) {
      const int Hc = Hl[l], Wc = Hl[l];
      const float* src; int Cis, cin;
      if (l == 3) { src = E[3]; Cis = CsE[3]; cin = 384; }
      else {
        long tot = S[l] * (long)CsCat[l];
        k_concat<<<pwb(tot), 256, 0, stream>>>(E[l], Rn[l + 1], Tcat, Hc, Wc, CsE[l], CsR[l + 1],
                                               2 * RCc[l], RCc[l + 1], CsCat[l], (int)tot);
        src = Tcat; Cis = CsCat[l]; cin = catCin[l];
      }
      conv(src,  T1, wb_ci[l],   ci_b[l], nullptr, Hc, Wc, Cis,    cin,    RCc[l], 1, EP_LIN);
      conv(R[l], T2, wb_u0[l],   u0_b[l], R[l],    Hc, Wc, CsR[l], RCc[l], RCc[l], 1, EP_SIGMUL);
      conv(T2,   T3, wb_winh[l], nullptr, nullptr, Hc, Wc, CsR[l], RCc[l], RCc[l], 9, EP_LIN);
      {
        long tot = S[l] * (long)CsR[l];
        k_supp<<<pwb(tot), 256, 0, stream>>>(T1, T3, R[l], alpha[l], mu[l], T4, (int)tot, CsR[l], RCc[l]);
      }
      conv(T4, T2, wb_u1[l],   u1_b[l], nullptr, Hc, Wc, CsR[l], RCc[l], RCc[l], 1, EP_SIG);
      conv(T4, T3, wb_wexc[l], nullptr, nullptr, Hc, Wc, CsR[l], RCc[l], RCc[l], 9, EP_SP);
      {
        long tot = S[l] * (long)CsR[l];
        k_combine<<<pwb(tot), 256, 0, stream>>>(T4, T3, T2, R[l], kappa[l], wwp[l], Rn[l],
                                                (int)tot, CsR[l], RCc[l]);
      }
    }
    // bottom-up
    { long tot = S[0] * 16L; k_xcvt<<<pwb(tot), 256, 0, stream>>>(x, TA, t, (int)tot); }
    for (int l = 0; l < 4; ++l) {
      const int Hc = Hl[l], Wc = Hl[l];
      conv(Rn[l], TH, wb_conv[l], conv_b[l], nullptr, Hc, Wc, CsR[l], RCc[l], RCc[l], 9,
           l == 0 ? EP_RELUCLIP : EP_RELU);
      {
        int HW = Hc * Wc;
        long tot = (long)HW * CsE[l];
        int bx = (int)((tot + 255) / 256); if (bx > 256) bx = 256; if (bx < 1) bx = 1;
        dim3 g(bx, 4);
        k_err<<<g, 256, 0, stream>>>(TH, TA, E[l], partial, HW, CsE[l], CsR[l], RCc[l], l == 0 ? 1 : 0);
        k_reduce_err<<<1, 4, 0, stream>>>(partial, bx, me + t * 16 + l, fe + t * 4, l == 0 ? 1 : 0);
      }
      if (l < 3) {
        conv(E[l], Tupd, wb_upd[l], upd_b[l], nullptr, Hc, Wc, CsE[l], 2 * RCc[l], RCc[l + 1], 9, EP_LIN);
        long tot = S[l + 1] * (long)CsR[l + 1];
        k_pool<<<pwb(tot), 256, 0, stream>>>(Tupd, TA, Hl[l + 1], Hl[l + 1], CsR[l + 1], (int)tot);
      }
    }
    for (int l = 0; l < 4; ++l) { float* tp = R[l]; R[l] = Rn[l]; Rn[l] = tp; }
  }

  k_finalize<<<1, 32, 0, stream>>>(me, fe, (float*)d_out);
}
